// MultiHeadAttention_53575422050472
// MI455X (gfx1250) — matrix operand order repaired and checked by reading, not compiled, not run on hardware
//
#include <hip/hip_runtime.h>
#include <hip/hip_bf16.h>

// ---------------------------------------------------------------------------
// Problem constants (from reference): B=4, S=256, HID=512, H=8, D=64
// ---------------------------------------------------------------------------
#define Bc   4
#define Sc   256
#define HIDc 512
#define Hc   8
#define Dc   64

typedef __attribute__((ext_vector_type(16))) __bf16 v16bf;
typedef __attribute__((ext_vector_type(8)))  __bf16 v8bf;
typedef __attribute__((ext_vector_type(8)))  float  v8f;

// concatenate two 8-element bf16 vectors into a WMMA operand (register moves)
static __device__ __forceinline__ v16bf cat8(v8bf lo, v8bf hi) {
    v16bf r;
#pragma unroll
    for (int e = 0; e < 8; ++e) { r[e] = lo[e]; r[e + 8] = hi[e]; }
    return r;
}

// A fragment (16x32 bf16 MxK): lanes 0-15 -> M=lane, K={k0..k0+7, k0+16..k0+23};
// lanes 16-31 -> M=lane-16, K={k0+8..k0+15, k0+24..k0+31}.
// rowbase = this lane's row, bf16, contiguous along K -> two b128 loads.
static __device__ __forceinline__ v16bf load_a_bf(const __bf16* __restrict__ rowbase,
                                                  int k0, int lane) {
    const int kb = (lane < 16) ? 0 : 8;
    return cat8(*(const v8bf*)(rowbase + k0 + kb),
                *(const v8bf*)(rowbase + k0 + 16 + kb));
}

// B fragment (32x16 bf16 KxN): lane n<16 -> column n, K={k0..k0+7, k0+16..k0+23};
// lane n+16 -> column n, K={k0+8..k0+15, k0+24..k0+31}.
// colbase = this lane's column, bf16, contiguous along K -> two b128 loads.
static __device__ __forceinline__ v16bf load_b_bf(const __bf16* __restrict__ colbase,
                                                  int k0, int lane) {
    const int kh_ = (lane < 16) ? 0 : 8;
    return cat8(*(const v8bf*)(colbase + k0 + kh_),
                *(const v8bf*)(colbase + k0 + 16 + kh_));
}

// ---------------------------------------------------------------------------
// Prep 1: convert q/k/v activations [1024x512] f32 -> bf16 (same layout).
// grid (256,3) x 256 threads, 8 elements/thread.
// ---------------------------------------------------------------------------
__global__ __launch_bounds__(256) void prep_x_kernel(
    const float* __restrict__ q, const float* __restrict__ k,
    const float* __restrict__ v, __bf16* __restrict__ xbf)
{
    const float* src = (blockIdx.y == 0) ? q : (blockIdx.y == 1) ? k : v;
    __bf16* dst = xbf + (size_t)blockIdx.y * (1024 * HIDc);
    const size_t i = ((size_t)blockIdx.x * 256 + threadIdx.x) * 8;
    const float4 a = *(const float4*)(src + i);
    const float4 b = *(const float4*)(src + i + 4);
    v8bf o;
    o[0] = (__bf16)a.x; o[1] = (__bf16)a.y; o[2] = (__bf16)a.z; o[3] = (__bf16)a.w;
    o[4] = (__bf16)b.x; o[5] = (__bf16)b.y; o[6] = (__bf16)b.z; o[7] = (__bf16)b.w;
    *(v8bf*)(dst + i) = o;
}

// ---------------------------------------------------------------------------
// Prep 2: transpose+convert weights: Wt[n][k] = bf16(W[k][n]).
// LDS-tiled 32x32 transpose. grid (16,16,4), block (32,8).
// ---------------------------------------------------------------------------
__global__ void prep_w_kernel(
    const float* __restrict__ Wq, const float* __restrict__ Wk,
    const float* __restrict__ Wv, const float* __restrict__ Wo,
    __bf16* __restrict__ wt)
{
    const float* W = (blockIdx.z == 0) ? Wq : (blockIdx.z == 1) ? Wk
                   : (blockIdx.z == 2) ? Wv : Wo;
    __bf16* Wt = wt + (size_t)blockIdx.z * (HIDc * HIDc);
    __shared__ float tile[32][33];
    const int x0 = blockIdx.x * 32, y0 = blockIdx.y * 32;
    for (int j = threadIdx.y; j < 32; j += 8)
        tile[j][threadIdx.x] = W[(size_t)(y0 + j) * HIDc + x0 + threadIdx.x];
    __syncthreads();
    for (int j = threadIdx.y; j < 32; j += 8)
        Wt[(size_t)(x0 + j) * HIDc + y0 + threadIdx.x] = (__bf16)tile[threadIdx.x][j];
}

// ---------------------------------------------------------------------------
// Kernel 1: fused QKV projection, all-bf16 operands, contiguous b128 fragments.
// Emits: qh f32 [B,H,S,D]; kh f32 [B,H,S,D]; khB bf16 [B,H,S,D];
//        vhT bf16 [B,H,D,S] (transposed so PV B-fragments are contiguous).
// ---------------------------------------------------------------------------
__global__ __launch_bounds__(256) void qkv_proj_kernel(
    const __bf16* __restrict__ xbf, const __bf16* __restrict__ wt,
    const float* __restrict__ bq, const float* __restrict__ bk,
    const float* __restrict__ bv,
    float* __restrict__ qh, float* __restrict__ kh,
    __bf16* __restrict__ khB, __bf16* __restrict__ vhT)
{
    const int which = blockIdx.y;
    const __bf16* X  = xbf + (size_t)which * (1024 * HIDc);
    const __bf16* Wt = wt  + (size_t)which * (HIDc * HIDc);
    const float* bias = (which == 0) ? bq : (which == 1) ? bk : bv;

    const int wave = threadIdx.x >> 5, lane = threadIdx.x & 31;
    const int tile = blockIdx.x * 8 + wave;        // 2048 tiles
    const int m0 = (tile >> 5) << 4;
    const int n0 = (tile & 31) << 4;

    const __bf16* arow = X  + (size_t)(m0 + (lane & 15)) * HIDc;
    const __bf16* brow = Wt + (size_t)(n0 + (lane & 15)) * HIDc;

    v8f acc = {};
#pragma unroll 4
    for (int k0 = 0; k0 < HIDc; k0 += 32) {
        v16bf a = load_a_bf(arow, k0, lane);
        v16bf b = load_b_bf(brow, k0, lane);
        acc = __builtin_amdgcn_wmma_f32_16x16x32_bf16(false, a, false, b,
                                                      (short)0, acc, false, false);
    }

    const int col = n0 + (lane & 15);
    const int h = col >> 6, d = col & 63;
    const float bb = bias[col];
    const int rbase = (lane < 16) ? 0 : 8;
#pragma unroll
    for (int j = 0; j < 8; ++j) {
        const int m = m0 + rbase + j;              // b*S + s
        const int b_ = m >> 8, s = m & 255;
        const int bh = b_ * Hc + h;
        const float val = acc[j] + bb;
        if (which == 0) {
            qh[((size_t)bh * Sc + s) * Dc + d] = val;
        } else if (which == 1) {
            kh[((size_t)bh * Sc + s) * Dc + d] = val;
            khB[((size_t)bh * Sc + s) * Dc + d] = (__bf16)val;
        } else {
            vhT[((size_t)bh * Dc + d) * Sc + s] = (__bf16)val;
        }
    }
}

// ---------------------------------------------------------------------------
// Kernel 2: attention core.  One block per (b, h, 32-row s-tile), 8 waves.
// ---------------------------------------------------------------------------
__global__ __launch_bounds__(256) void attn_kernel(
    const float* __restrict__ qh, const float* __restrict__ kh,
    const __bf16* __restrict__ khB, const __bf16* __restrict__ vhT,
    const float* __restrict__ attn_bias,
    const float* __restrict__ spq, const float* __restrict__ spk,
    __bf16* __restrict__ ctx)
{
    __shared__ float  sQ[32][Dc];      // 8 KB  (f32, for bias dot)
    __shared__ __bf16 sQbf[32][Dc];    // 4 KB  (bf16 shadow for WMMA A)
    __shared__ float  sS[32][Sc];      // 32 KB (scores)
    __shared__ __bf16 sP[32][Sc];      // 16 KB (probabilities, WMMA A)

    const int bh = blockIdx.x >> 3;              // b*H + h
    const int s0 = (blockIdx.x & 7) << 5;
    const int wave = threadIdx.x >> 5, lane = threadIdx.x & 31;

    const float*  qbase = qh  + ((size_t)bh * Sc + s0) * Dc;   // 32 x 64
    const float*  kbase = kh  + (size_t)bh * Sc * Dc;          // 256 x 64
    const __bf16* kBb   = khB + (size_t)bh * Sc * Dc;          // 256 x 64 bf16
    const __bf16* vTb   = vhT + (size_t)bh * Dc * Sc;          // 64 x 256 bf16

    // ---- stage Q tile (8 KB) via CDNA5 async global->LDS DMA ----
    {
        const unsigned lds0 = (unsigned)(size_t)(&sQ[0][0]);
#pragma unroll
        for (int pass = 0; pass < 2; ++pass) {
            const unsigned loff = lds0 + pass * 4096u + threadIdx.x * 16u;
            const unsigned long long gaddr =
                (unsigned long long)((const char*)qbase + pass * 4096 + threadIdx.x * 16);
            asm volatile("global_load_async_to_lds_b128 %0, %1, off"
                         :: "v"(loff), "v"(gaddr) : "memory");
        }
        asm volatile("s_wait_asynccnt 0x0" ::: "memory");
    }
    __syncthreads();
    for (int i = threadIdx.x; i < 32 * Dc; i += 256)
        sQbf[i >> 6][i & 63] = (__bf16)sQ[i >> 6][i & 63];
    __syncthreads();

    // ---- phase 1: raw QK^T scores via WMMA ----
    for (int tidx = wave; tidx < 32; tidx += 8) {
        const int si = tidx >> 4;
        const int t0 = (tidx & 15) << 4;
        const __bf16* arow = &sQbf[si * 16 + (lane & 15)][0];
        const __bf16* bcol = kBb + (size_t)(t0 + (lane & 15)) * Dc; // contiguous in d
        v8f acc = {};
#pragma unroll
        for (int k0 = 0; k0 < Dc; k0 += 32) {
            v16bf a = load_a_bf(arow, k0, lane);
            v16bf b = load_b_bf(bcol, k0, lane);
            acc = __builtin_amdgcn_wmma_f32_16x16x32_bf16(false, a, false, b,
                                                          (short)0, acc, false, false);
        }
        const int rbase = (lane < 16) ? 0 : 8;
#pragma unroll
        for (int j = 0; j < 8; ++j)
            sS[si * 16 + rbase + j][t0 + (lane & 15)] = acc[j];
    }
    __syncthreads();

    // ---- phase 2: GRPE bias stream (roofline term: 1.07 GB of spq/spk) ----
    // lanes 0-15 -> t, lanes 16-31 -> t+1; float4/lane => 512 contiguous
    // bytes per wave load; 4-step half-wave shfl reduction; prefetch ahead.
    const float scale = 0.125f;                  // D^-0.5
    const float* abias = attn_bias + ((size_t)bh * Sc + s0) * Sc;
    const int lh = lane >> 4;                    // which of the two t's
    const int ll = lane & 15;                    // d quarter (4 floats each)
    const size_t srs = (size_t)Sc * Dc;          // s-row stride in spq/spk
    for (int t2 = wave * 2; t2 < Sc; t2 += 16) {
        const int t = t2 + lh;
        const float4 kv = *(const float4*)(kbase + (size_t)t * Dc + ll * 4);
        const float* pq_p = spq + (((size_t)bh * Sc + s0) * Sc + t) * Dc + ll * 4;
        const float* pk_p = spk + (((size_t)bh * Sc + s0) * Sc + t) * Dc + ll * 4;
#pragma unroll 4
        for (int srow = 0; srow < 32; ++srow) {
            const float4 pq = *(const float4*)(pq_p + (size_t)srow * srs);
            const float4 pk = *(const float4*)(pk_p + (size_t)srow * srs);
            if (srow + 1 < 32) {
                __builtin_prefetch(pq_p + (size_t)(srow + 1) * srs, 0, 1);
                __builtin_prefetch(pk_p + (size_t)(srow + 1) * srs, 0, 1);
            }
            const float4 qv = *(const float4*)(&sQ[srow][ll * 4]);
            float part = qv.x * pq.x + qv.y * pq.y + qv.z * pq.z + qv.w * pq.w
                       + kv.x * pk.x + kv.y * pk.y + kv.z * pk.z + kv.w * pk.w;
            part += __shfl_xor(part, 8, 32);
            part += __shfl_xor(part, 4, 32);
            part += __shfl_xor(part, 2, 32);
            part += __shfl_xor(part, 1, 32);
            if (ll == 0)
                sS[srow][t] = (sS[srow][t] + part) * scale + abias[(size_t)srow * Sc + t];
        }
    }
    __syncthreads();

    // ---- phase 3: softmax; write probabilities as bf16 WMMA A operand ----
    for (int r = wave; r < 32; r += 8) {
        float vals[8];
        float m = -1e30f;
#pragma unroll
        for (int i = 0; i < 8; ++i) { vals[i] = sS[r][lane + i * 32]; m = fmaxf(m, vals[i]); }
        m = fmaxf(m, __shfl_xor(m, 16, 32));
        m = fmaxf(m, __shfl_xor(m,  8, 32));
        m = fmaxf(m, __shfl_xor(m,  4, 32));
        m = fmaxf(m, __shfl_xor(m,  2, 32));
        m = fmaxf(m, __shfl_xor(m,  1, 32));
        float sum = 0.f;
#pragma unroll
        for (int i = 0; i < 8; ++i) { vals[i] = __expf(vals[i] - m); sum += vals[i]; }
        sum += __shfl_xor(sum, 16, 32);
        sum += __shfl_xor(sum,  8, 32);
        sum += __shfl_xor(sum,  4, 32);
        sum += __shfl_xor(sum,  2, 32);
        sum += __shfl_xor(sum,  1, 32);
        const float inv = 1.0f / sum;
#pragma unroll
        for (int i = 0; i < 8; ++i) sP[r][lane + i * 32] = (__bf16)(vals[i] * inv);
    }
    __syncthreads();

    // ---- phase 4: P @ V, one 16x16 tile per wave; fused transpose out ----
    {
        const int si = wave >> 2;                // 0..1
        const int dj = wave & 3;                 // 0..3 -> d tile
        const __bf16* arow = &sP[si * 16 + (lane & 15)][0];
        const __bf16* bcol = vTb + (size_t)(dj * 16 + (lane & 15)) * Sc; // contiguous in t
        v8f acc = {};
#pragma unroll 2
        for (int k0 = 0; k0 < Sc; k0 += 32) {
            v16bf a = load_a_bf(arow, k0, lane);
            v16bf b = load_b_bf(bcol, k0, lane);
            acc = __builtin_amdgcn_wmma_f32_16x16x32_bf16(false, a, false, b,
                                                          (short)0, acc, false, false);
        }
        const int b_ = bh >> 3, h = bh & 7;
        const int col = h * Dc + dj * 16 + (lane & 15);
        const int rbase = (lane < 16) ? 0 : 8;
#pragma unroll
        for (int j = 0; j < 8; ++j) {
            const int s = s0 + si * 16 + rbase + j;
            ctx[((size_t)(b_ * Sc + s)) * HIDc + col] = (__bf16)acc[j];
        }
    }
}

// ---------------------------------------------------------------------------
// Kernel 3: output projection. out = ctx[1024x512]bf16 @ WoT + bo -> f32
// ---------------------------------------------------------------------------
__global__ __launch_bounds__(256) void out_proj_kernel(
    const __bf16* __restrict__ ctx, const __bf16* __restrict__ WoT,
    const float* __restrict__ bias, float* __restrict__ out)
{
    const int wave = threadIdx.x >> 5, lane = threadIdx.x & 31;
    const int tile = blockIdx.x * 8 + wave;
    const int m0 = (tile >> 5) << 4;
    const int n0 = (tile & 31) << 4;

    const __bf16* arow = ctx + (size_t)(m0 + (lane & 15)) * HIDc;
    const __bf16* brow = WoT + (size_t)(n0 + (lane & 15)) * HIDc;

    v8f acc = {};
#pragma unroll 4
    for (int k0 = 0; k0 < HIDc; k0 += 32) {
        v16bf a = load_a_bf(arow, k0, lane);
        v16bf b = load_b_bf(brow, k0, lane);
        acc = __builtin_amdgcn_wmma_f32_16x16x32_bf16(false, a, false, b,
                                                      (short)0, acc, false, false);
    }

    const int col = n0 + (lane & 15);
    const float bb = bias[col];
    const int rbase = (lane < 16) ? 0 : 8;
#pragma unroll
    for (int j = 0; j < 8; ++j)
        out[(size_t)(m0 + rbase + j) * HIDc + col] = acc[j] + bb;
}

// ---------------------------------------------------------------------------
// Launch
// ---------------------------------------------------------------------------
extern "C" void kernel_launch(void* const* d_in, const int* in_sizes, int n_in,
                              void* d_out, int out_size, void* d_ws, size_t ws_size,
                              hipStream_t stream) {
    const float* q         = (const float*)d_in[0];
    const float* k         = (const float*)d_in[1];
    const float* v         = (const float*)d_in[2];
    const float* attn_bias = (const float*)d_in[3];
    const float* spq       = (const float*)d_in[4];
    const float* spk       = (const float*)d_in[5];
    const float* Wq        = (const float*)d_in[6];
    const float* bq        = (const float*)d_in[7];
    const float* Wk        = (const float*)d_in[8];
    const float* bk        = (const float*)d_in[9];
    const float* Wv        = (const float*)d_in[10];
    const float* bv        = (const float*)d_in[11];
    const float* Wo        = (const float*)d_in[12];
    const float* bo        = (const float*)d_in[13];

    // Workspace layout (12 MB total):
    //   [0,2M)   qh   f32  [B,H,S,D]
    //   [2M,4M)  kh   f32  [B,H,S,D]
    //   [4M,5M)  khB  bf16 [B,H,S,D]
    //   [5M,6M)  vhT  bf16 [B,H,D,S]
    //   [6M,7M)  ctx  bf16 [B,S,HID]
    //   [7M,10M) xbf  bf16 3 x [1024,512]
    //   [10M,12M) wt  bf16 4 x [512,512] (transposed weights)
    char* w = (char*)d_ws;
    float*  qh  = (float*)(w);
    float*  kh  = (float*)(w + (2u << 20));
    __bf16* khB = (__bf16*)(w + (4u << 20));
    __bf16* vhT = (__bf16*)(w + (5u << 20));
    __bf16* ctx = (__bf16*)(w + (6u << 20));
    __bf16* xbf = (__bf16*)(w + (7u << 20));
    __bf16* wt  = (__bf16*)(w + (10u << 20));

    prep_x_kernel<<<dim3(256, 3), 256, 0, stream>>>(q, k, v, xbf);
    prep_w_kernel<<<dim3(16, 16, 4), dim3(32, 8), 0, stream>>>(Wq, Wk, Wv, Wo, wt);
    qkv_proj_kernel<<<dim3(256, 3), 256, 0, stream>>>(xbf, wt, bq, bk, bv,
                                                      qh, kh, khB, vhT);
    attn_kernel<<<256, 256, 0, stream>>>(qh, kh, khB, vhT, attn_bias, spq, spk, ctx);
    out_proj_kernel<<<256, 256, 0, stream>>>(ctx, wt + 3 * (size_t)(HIDc * HIDc), bo,
                                             (float*)d_out);
}
